// unit_gcn_54700703482538
// MI455X (gfx1250) — matrix operand order repaired and checked by reading, not compiled, not run on hardware
//
#include <hip/hip_runtime.h>

typedef __attribute__((ext_vector_type(16))) __bf16 v16bf;
typedef __attribute__((ext_vector_type(8)))  __bf16 v8bf;
typedef __attribute__((ext_vector_type(8)))  float  v8f;

#define N_   64
#define C_   64
#define T_   300
#define V_   25
#define O_   64
#define KTOT 192   // 3 * C_
#define TSUB 8     // t's per block
#define JCOL 200   // TSUB * V_
#define JPAD 208   // 13 n-tiles of 16
#define KPAD 208   // padded K pitch (16B aligned rows, spread banks)

__device__ __forceinline__ v16bf cat16(v8bf lo, v8bf hi) {
  return __builtin_shufflevector(lo, hi, 0,1,2,3,4,5,6,7,8,9,10,11,12,13,14,15);
}

// ---------------------------------------------------------------------------
// Kernel 1: fused  z = (x @ A) ; y = Wcat @ z   for one (n, t-tile) per block.
// All operands staged in LDS as zero-padded bf16 so every WMMA fragment is an
// aligned ds_load_b128 pair; no divergence or global latency in the MMA loops.
// ---------------------------------------------------------------------------
__global__ __launch_bounds__(256)
void gcn_fused(const float* __restrict__ x, const float* __restrict__ Ag,
               const float* __restrict__ Wg, float* __restrict__ y)
{
  __shared__ __bf16 sAT[3][32][32];        // A^T: [a][w][v], zero padded
  __shared__ __bf16 sX[C_][TSUB][32];      // x tile: [c][t][v], zero padded
  __shared__ __bf16 sW[O_][KPAD];          // Wcat: [o][a*64+c]
  __shared__ __bf16 sZ[JPAD][KPAD];        // z^T : [j=t*25+w][k=a*64+c]

  const int tid  = threadIdx.x;
  const int lane = tid & 31;
  const int wave = tid >> 5;
  const int row  = lane & 15;
  const int half = lane >> 4;              // 0: lanes 0-15, 1: lanes 16-31
  const int n    = blockIdx.y;
  const int t0   = blockIdx.x * TSUB;

  // --- stage A^T (zero pad: K-dim padding multiplies to exact zero) -------
  for (int i = tid; i < 3*32*32; i += 256) {
    int a = i >> 10, r = i & 1023, w = r >> 5, v = r & 31;
    float val = (w < V_ && v < V_) ? Ag[a*(V_*V_) + v*V_ + w] : 0.f;
    sAT[a][w][v] = (__bf16)val;
  }
  // --- stage x tile, coalesced (consecutive tid -> consecutive v) ---------
  for (int i = tid; i < C_*TSUB*32; i += 256) {
    int c = i >> 8, t = (i >> 5) & 7, v = i & 31;
    float val = 0.f;
    if (v < V_ && (t0 + t) < T_)
      val = x[(((size_t)n*C_ + c)*T_ + (t0 + t))*V_ + v];
    sX[c][t][v] = (__bf16)val;
  }
  // --- stage Wcat[o][a*64+c] = W[a][o][c] ---------------------------------
  for (int i = tid; i < O_*KTOT; i += 256) {
    int o = i / KTOT, k = i % KTOT, a = k >> 6, c = k & 63;
    sW[o][k] = (__bf16)Wg[((size_t)a*O_ + o)*C_ + c];
  }
  __syncthreads();

  // --- GEMM1: per (t, a, ctile): z[c][w] = X_t(64x32) @ A_a(32x32) --------
  for (int task = wave; task < TSUB*3*4; task += 8) {
    int tl = task / 12, rem = task % 12, a = rem >> 2, ct = rem & 3;
    // A-fragment: row c = ct*16+row, K = v (padding already zeroed in LDS)
    const __bf16* xr = &sX[ct*16 + row][tl][0];
    v8bf x0 = *(const v8bf*)(xr + (half ? 8  : 0));
    v8bf x1 = *(const v8bf*)(xr + (half ? 24 : 16));
    v16bf xa = cat16(x0, x1);
    #pragma unroll
    for (int wt = 0; wt < 2; ++wt) {
      int w = wt*16 + row;                 // B-fragment column for this lane
      const v8bf* bp = (const v8bf*)&sAT[a][w][half ? 8 : 0];
      v16bf bb = cat16(bp[0], bp[2]);
      v8f zero = {};
      v8f d = __builtin_amdgcn_wmma_f32_16x16x32_bf16(
                  false, xa, false, bb, (short)0, zero, false, false);
      if (w < V_) {                        // drop padded output columns
        int j  = tl*V_ + w;
        int kb = a*64 + ct*16 + (half ? 8 : 0);
        #pragma unroll
        for (int r = 0; r < 8; ++r) sZ[j][kb + r] = (__bf16)d[r];
      }
    }
  }
  __syncthreads();

  // --- GEMM2: y[o][j] = sum_k Wcat[o][k] * z[k][j],  K = 192 --------------
  for (int task = wave; task < 4*13; task += 8) {
    int mt = task / 13, nt = task % 13;
    int j  = nt*16 + row;
    v8f acc = {};
    #pragma unroll
    for (int ks = 0; ks < 6; ++ks) {
      int k0 = ks*32;
      const __bf16* wr = &sW[mt*16 + row][k0];
      v8bf a0 = *(const v8bf*)(wr + (half ? 8  : 0));
      v8bf a1 = *(const v8bf*)(wr + (half ? 24 : 16));
      v16bf af = cat16(a0, a1);
      const v8bf* zp = (const v8bf*)&sZ[j][k0 + (half ? 8 : 0)];
      v16bf bf_ = cat16(zp[0], zp[2]);
      acc = __builtin_amdgcn_wmma_f32_16x16x32_bf16(
                false, af, false, bf_, (short)0, acc, false, false);
    }
    if (j < JCOL) {
      int t = t0 + j / V_, w = j % V_;
      if (t < T_) {
        #pragma unroll
        for (int r = 0; r < 8; ++r) {
          int o = mt*16 + r + (half ? 8 : 0);
          y[(((size_t)n*O_ + o)*T_ + t)*V_ + w] = acc[r];
        }
      }
    }
  }
}

// ---------------------------------------------------------------------------
// Kernel 2: deterministic per-channel batch stats -> scale/bias in d_ws.
// One block per channel, fixed-order tree reduction, float4 streaming.
// (T_*V_ = 7500 is a multiple of 4, so float4 never straddles a channel.)
// ---------------------------------------------------------------------------
__global__ __launch_bounds__(256)
void bn_stats(const float* __restrict__ y, const float* __restrict__ gamma,
              const float* __restrict__ beta, float2* __restrict__ sb)
{
  const int o = blockIdx.x, tid = threadIdx.x;
  float s = 0.f, q = 0.f;
  for (int n = 0; n < N_; ++n) {
    const float4* base = (const float4*)(y + ((size_t)n*O_ + o)*(T_*V_));
    for (int i = tid; i < (T_*V_)/4; i += 256) {
      float4 v = base[i];
      s += v.x + v.y + v.z + v.w;
      q += v.x*v.x + v.y*v.y + v.z*v.z + v.w*v.w;
    }
  }
  __shared__ float rs[256], rq[256];
  rs[tid] = s; rq[tid] = q;
  __syncthreads();
  for (int off = 128; off > 0; off >>= 1) {
    if (tid < off) { rs[tid] += rs[tid+off]; rq[tid] += rq[tid+off]; }
    __syncthreads();
  }
  if (tid == 0) {
    const float inv = 1.f / (float)(N_*T_*V_);
    float mean = rs[0]*inv;
    float var  = rq[0]*inv - mean*mean;          // biased var, matches jnp.var
    float sc   = gamma[o] * rsqrtf(var + 1e-5f);
    sb[o] = make_float2(sc, beta[o] - mean*sc);
  }
}

// ---------------------------------------------------------------------------
// Kernel 3: in-place y = relu(y*scale + bias), float4 streaming.
// ---------------------------------------------------------------------------
__global__ __launch_bounds__(256)
void bn_apply(float4* __restrict__ y4, const float2* __restrict__ sb)
{
  size_t idx = (size_t)blockIdx.x*256 + threadIdx.x;       // float4 index
  int o = (int)((idx / (size_t)((T_*V_)/4)) & (O_ - 1));
  float2 p = sb[o];
  float4 v = y4[idx];
  v.x = fmaxf(v.x*p.x + p.y, 0.f);
  v.y = fmaxf(v.y*p.x + p.y, 0.f);
  v.z = fmaxf(v.z*p.x + p.y, 0.f);
  v.w = fmaxf(v.w*p.x + p.y, 0.f);
  y4[idx] = v;
}

extern "C" void kernel_launch(void* const* d_in, const int* in_sizes, int n_in,
                              void* d_out, int out_size, void* d_ws, size_t ws_size,
                              hipStream_t stream)
{
  (void)in_sizes; (void)n_in; (void)out_size; (void)ws_size;
  const float* x = (const float*)d_in[0];
  const float* A = (const float*)d_in[1];
  const float* W = (const float*)d_in[2];
  const float* g = (const float*)d_in[3];
  const float* b = (const float*)d_in[4];
  float*  y  = (float*)d_out;
  float2* sb = (float2*)d_ws;              // 64 * 8 B scratch

  dim3 grid1((T_ + TSUB - 1)/TSUB, N_);    // 38 x 64 blocks
  gcn_fused<<<grid1, 256, 0, stream>>>(x, A, W, y);
  bn_stats<<<O_, 256, 0, stream>>>(y, g, b, sb);
  size_t total4 = (size_t)N_*O_*T_*V_/4;   // 7,680,000 (multiple of 256)
  bn_apply<<<(unsigned)(total4/256), 256, 0, stream>>>((float4*)y, sb);
}
